// DeepSeekV3Block_45500883533881
// MI455X (gfx1250) — matrix operand order repaired and checked by reading, not compiled, not run on hardware
//
#include <hip/hip_runtime.h>
#include <math.h>

typedef __bf16 bf16;
typedef __attribute__((ext_vector_type(16))) __bf16 v16bf;
typedef __attribute__((ext_vector_type(8)))  float  v8f;

#define NT     2048   // B*S tokens
#define DMODEL 2048
#define RLAT   512
#define NH     16
#define DH     128
#define NE     8
#define HID    1024
#define SEQ    1024
#define BATCH  2
#define CAP    2048   // worst-case tokens per expert

#if __has_builtin(__builtin_amdgcn_tensor_load_to_lds) && \
    __has_builtin(__builtin_amdgcn_s_wait_tensorcnt)
#define USE_TDM 1
#else
#define USE_TDM 0
#endif

// ---------------- WMMA helpers (CDNA5 16x16x32 bf16, fp32 accum) ----------------

__device__ __forceinline__ v8f v8f_zero() {
  v8f z;
#pragma unroll
  for (int i = 0; i < 8; ++i) z[i] = 0.f;
  return z;
}

__device__ __forceinline__ v8f wmma_bf16(v16bf a, v16bf b, v8f c) {
  return __builtin_amdgcn_wmma_f32_16x16x32_bf16(false, a, false, b, (short)0, c,
                                                 false, false);
}

// A fragment (16x32 bf16), row-major source with leading dim lda.
// lanes 0-15: row=lane, K=0..15; lanes 16-31: same rows, K=16..31.
// Per lane this is one contiguous 32B run -> compiler merges into b128 loads.
__device__ __forceinline__ v16bf load_afrag(const bf16* A, int lda, int lane) {
  const int row = lane & 15;
  const int kh  = (lane >> 4) * 16;
  const bf16* p = A + (size_t)row * lda;
  v16bf a;
#pragma unroll
  for (int i = 0; i < 8; ++i) {
    int k = i * 2 + kh;
    a[2 * i]     = p[k];
    a[2 * i + 1] = p[k + 1];
  }
  return a;
}

// B fragment (32x16) where this lane's 16 K-elements are contiguous in memory
// (caller applies the column offset): lanes 0-15 K=0..15, lanes 16-31 K=16..31.
__device__ __forceinline__ v16bf load_bfrag_rowk(const bf16* Brow, int lane) {
  const int kb = (lane >> 4) * 16;
  v16bf b;
#pragma unroll
  for (int i = 0; i < 16; ++i) b[i] = Brow[kb + i];
  return b;
}

// ---------------- Tensor Data Mover: 2-D bf16 tile -> LDS -------------------------

#if USE_TDM
typedef unsigned int tdm_u32x4 __attribute__((ext_vector_type(4)));
typedef int          tdm_i32x8 __attribute__((ext_vector_type(8)));
typedef int          tdm_i32x4 __attribute__((ext_vector_type(4)));

__device__ __forceinline__ unsigned lds_offset_of(const void* p) {
  return (unsigned)(size_t)(__attribute__((address_space(3))) const void*)p;
}

// Load a tile_h x tile_w (elements, 2B each) tile, row stride `stride` elements,
// into LDS at byte offset lds_off. OOB rows (>= tensor_h) read as zero.
__device__ __forceinline__ void tdm_load_2d_bf16(unsigned lds_off, const void* gaddr,
                                                 unsigned tile_w, unsigned tile_h,
                                                 unsigned tensor_w, unsigned tensor_h,
                                                 unsigned stride) {
  const unsigned long long ga = (unsigned long long)(size_t)gaddr;
  tdm_u32x4 g0;
  g0[0] = 1u;                                   // count=1, user descriptor
  g0[1] = lds_off;                              // lds_addr (bytes)
  g0[2] = (unsigned)ga;                         // global_addr[31:0]
  g0[3] = (unsigned)((ga >> 32) & 0x01FFFFFFu)  // global_addr[56:32]
          | (2u << 30);                         // type = 2 ("image")
  tdm_i32x8 g1;
  g1[0] = (int)(1u << 16);                      // data_size=1 (2 bytes); mask=0
  g1[1] = (int)((tensor_w & 0xFFFFu) << 16);    // tensor_dim0[15:0]
  g1[2] = (int)(((tensor_w >> 16) & 0xFFFFu) | ((tensor_h & 0xFFFFu) << 16));
  g1[3] = (int)(((tensor_h >> 16) & 0xFFFFu) | ((tile_w & 0xFFFFu) << 16));
  g1[4] = (int)(tile_h & 0xFFFFu);              // tile_dim1; tile_dim2=0 (2D)
  g1[5] = (int)stride;                          // tensor_dim0_stride[31:0]
  g1[6] = 0;                                    // stride[47:32] | dim1_stride lo
  g1[7] = 0;
  tdm_i32x4 gz;
  gz[0] = 0; gz[1] = 0; gz[2] = 0; gz[3] = 0;
  tdm_i32x8 gz8;
#pragma unroll
  for (int i = 0; i < 8; ++i) gz8[i] = 0;
  // 6-arg form (clang-23 / therock-10.0 headers)
  __builtin_amdgcn_tensor_load_to_lds(g0, g1, gz, gz, gz8, 0);
}
#endif

// ---------------- Generic WMMA GEMM: C[MxN] = A[MxK](bf16) * B[KxN](f32->bf16) ---
// Block tile 128x64x32, 256 threads = 8 waves (4x2), wave tile 32x32.
// A tile staged by TDM (contiguous rows) or VGPRs (gathered rows).
// B tile converted f32->bf16 and stored TRANSPOSED [n][k] so fragment reads are
// two ds_load_b128 per fragment.

template <bool OUT_BF16, bool SILU, bool GATHER_A, bool SCATTER_ADD>
__global__ __launch_bounds__(256) void gemm_wmma(
    const bf16* __restrict__ A, const float* __restrict__ Bw, void* __restrict__ C,
    int M, int N, int Kd, int lda, int ldc, const int* __restrict__ rowmap,
    const float* __restrict__ rowscale, const int* __restrict__ counts) {
  const int e    = blockIdx.z;
  const int Meff = counts ? counts[e] : M;
  const int m0   = blockIdx.x * 128;
  if (m0 >= Meff) return;
  const int n0   = blockIdx.y * 64;
  const int tid  = threadIdx.x;
  const int lane = tid & 31;
  const int wave = tid >> 5;
  const int wm = wave >> 1, wn = wave & 1;
  const int hf = lane >> 4, ln = lane & 15;

  __shared__ __align__(16) bf16 As[128 * 32];
  __shared__ __align__(16) bf16 Bst[64 * 32];   // transposed: [n][k]

  const float* Bwe = Bw + (size_t)e * Kd * N;

  v8f acc[2][2];
#pragma unroll
  for (int i = 0; i < 2; ++i)
#pragma unroll
    for (int j = 0; j < 2; ++j) acc[i][j] = v8f_zero();

  bool manualA = GATHER_A;
#if !USE_TDM
  manualA = true;
#endif

  // manual A-tile loader assignment: 2 threads per row, 16 bf16 each
  const int ar = tid >> 1;
  const int ac = (tid & 1) * 16;
  const int agrow   = m0 + ar;
  const bool avalid = agrow < Meff;
  size_t asrc;
  if (GATHER_A)
    asrc = avalid ? (size_t)rowmap[(size_t)e * CAP + agrow] : 0;
  else
    asrc = (counts ? (size_t)e * CAP : (size_t)0) + (size_t)agrow;

  // B-tile loader: 8 threads per K-row, 8 floats each
  const int br = tid >> 3;
  const int bc = (tid & 7) * 8;

  for (int kk = 0; kk < Kd; kk += 32) {
    __syncthreads();
#if USE_TDM
    if (!GATHER_A && wave == 0) {
      const bf16* abase =
          A + ((counts ? (size_t)e * CAP : (size_t)0) + (size_t)m0) * lda + kk;
      tdm_load_2d_bf16(lds_offset_of(As), abase, /*tile_w=*/32, /*tile_h=*/128,
                       /*tensor_w=*/32, /*tensor_h=*/(unsigned)(Meff - m0),
                       /*stride=*/(unsigned)lda);
    }
#endif
    if (manualA) {
      const bf16* ap = A + asrc * (size_t)lda + kk + ac;
      uint4 v0, v1;
      if (avalid) {
        v0 = ((const uint4*)ap)[0];
        v1 = ((const uint4*)ap)[1];
      } else {
        v0 = make_uint4(0u, 0u, 0u, 0u);
        v1 = v0;
      }
      *((uint4*)(As + ar * 32 + ac))     = v0;
      *((uint4*)(As + ar * 32 + ac + 8)) = v1;
    }
    {  // B tile: f32 -> bf16, transposed store
      const float* bp = Bwe + (size_t)(kk + br) * N + n0 + bc;
      float4 f0 = ((const float4*)bp)[0];
      float4 f1 = ((const float4*)bp)[1];
      bf16 t[8];
      t[0] = (bf16)f0.x; t[1] = (bf16)f0.y; t[2] = (bf16)f0.z; t[3] = (bf16)f0.w;
      t[4] = (bf16)f1.x; t[5] = (bf16)f1.y; t[6] = (bf16)f1.z; t[7] = (bf16)f1.w;
#pragma unroll
      for (int i = 0; i < 8; ++i) Bst[(bc + i) * 32 + br] = t[i];
    }
#if USE_TDM
    if (!GATHER_A && wave == 0) __builtin_amdgcn_s_wait_tensorcnt(0);
#endif
    __syncthreads();

    v16bf a0 = load_afrag(As + (wm * 32) * 32, 32, lane);
    v16bf a1 = load_afrag(As + (wm * 32 + 16) * 32, 32, lane);
    v16bf b0 = load_bfrag_rowk(Bst + (wn * 32 + ln) * 32, lane);
    v16bf b1 = load_bfrag_rowk(Bst + (wn * 32 + 16 + ln) * 32, lane);
    acc[0][0] = wmma_bf16(a0, b0, acc[0][0]);
    acc[0][1] = wmma_bf16(a0, b1, acc[0][1]);
    acc[1][0] = wmma_bf16(a1, b0, acc[1][0]);
    acc[1][1] = wmma_bf16(a1, b1, acc[1][1]);
  }

  // Epilogue. C layout: VGPR r -> row (r + 8*(lane>=16)), col = lane&15.
#pragma unroll
  for (int i = 0; i < 2; ++i)
#pragma unroll
    for (int j = 0; j < 2; ++j)
#pragma unroll
      for (int r = 0; r < 8; ++r) {
        const int row = m0 + wm * 32 + i * 16 + r + 8 * hf;
        if (row >= Meff) continue;
        const int col = n0 + wn * 32 + j * 16 + ln;
        float val = acc[i][j][r];
        if (SILU) val = val / (1.f + __expf(-val));
        if (SCATTER_ADD) {
          const int   orow = rowmap[(size_t)e * CAP + row];
          const float g    = rowscale[(size_t)e * CAP + row];
          atomicAdd((float*)C + (size_t)orow * ldc + col, val * g);
        } else {
          const size_t crow = counts ? ((size_t)e * CAP + row) : (size_t)row;
          if (OUT_BF16)
            ((bf16*)C)[crow * ldc + col] = (bf16)val;
          else
            ((float*)C)[crow * ldc + col] = val;
        }
      }
}

// ---------------- Flash attention (causal, per-head) with WMMA --------------------
// grid = (SEQ/64, BATCH*NH); 128 threads = 4 waves, each wave owns 16 query rows.
// K tile staged by TDM (row-major); V tile staged transposed [dh][key] in VGPRs.

__global__ __launch_bounds__(128) void attn_wmma(const bf16* __restrict__ q,
                                                 const bf16* __restrict__ k,
                                                 const bf16* __restrict__ v,
                                                 bf16* __restrict__ o) {
  const int bh = blockIdx.y;
  const int b  = bh / NH;
  const int h  = bh % NH;
  const int q0 = blockIdx.x * 64;
  const int tid  = threadIdx.x;
  const int wave = tid >> 5;
  const int lane = tid & 31;
  const int hf = lane >> 4, ln = lane & 15;
  const int qrow = q0 + wave * 16;

  __shared__ __align__(16) bf16 Kt[32 * DH];    // [key][dh]
  __shared__ __align__(16) bf16 Vtt[DH * 32];   // transposed: [dh][key]
  __shared__ __align__(16) bf16 Pb[4][16 * 32];

  // Q fragments for this wave: 16 rows x 128 (4 chunks of K=32), loaded once.
  const bf16* qg = q + ((size_t)(b * SEQ + qrow)) * DMODEL + h * DH;
  v16bf aq[4];
#pragma unroll
  for (int c = 0; c < 4; ++c) aq[c] = load_afrag(qg + c * 32, DMODEL, lane);

  v8f oa[8];
#pragma unroll
  for (int s = 0; s < 8; ++s) oa[s] = v8f_zero();
  float mrow[8], lrow[8];
#pragma unroll
  for (int r = 0; r < 8; ++r) { mrow[r] = -3.0e38f; lrow[r] = 0.f; }
  const float scale = 0.08838834764831845f;  // 1/sqrt(128)

  for (int kb = 0; kb < q0 + 64; kb += 32) {
    __syncthreads();
#if USE_TDM
    if (wave == 0)
      tdm_load_2d_bf16(lds_offset_of(Kt),
                       k + ((size_t)(b * SEQ + kb)) * DMODEL + h * DH,
                       /*tile_w=*/DH, /*tile_h=*/32, /*tensor_w=*/DH,
                       /*tensor_h=*/32, /*stride=*/DMODEL);
#else
    {  // manual K tile: 32 keys x 128 dh
      const int r = tid >> 2;
      const int c = (tid & 3) * 32;
      const uint4* ks =
          (const uint4*)(k + ((size_t)(b * SEQ + kb + r)) * DMODEL + h * DH + c);
      uint4* kd = (uint4*)(Kt + r * DH + c);
#pragma unroll
      for (int i = 0; i < 4; ++i) kd[i] = ks[i];
    }
#endif
    {  // V tile transposed: [dh][key]
      const int r = tid >> 2;         // key 0..31
      const int c = (tid & 3) * 32;   // dh base
      const uint4* vs =
          (const uint4*)(v + ((size_t)(b * SEQ + kb + r)) * DMODEL + h * DH + c);
#pragma unroll
      for (int i = 0; i < 4; ++i) {
        uint4 raw = vs[i];
        bf16 t[8];
        *(uint4*)t = raw;
#pragma unroll
        for (int j = 0; j < 8; ++j) Vtt[(c + i * 8 + j) * 32 + r] = t[j];
      }
    }
#if USE_TDM
    if (wave == 0) __builtin_amdgcn_s_wait_tensorcnt(0);
#endif
    __syncthreads();

    // S = Q * K^T : two 16x16 score tiles (keys kb..+15, kb+16..+31)
    v8f s0 = v8f_zero(), s1 = v8f_zero();
#pragma unroll
    for (int c = 0; c < 4; ++c) {
      const bf16* kr0 = Kt + (size_t)ln * DH + c * 32;         // key = kb+ln
      const bf16* kr1 = Kt + (size_t)(16 + ln) * DH + c * 32;  // key = kb+16+ln
      s0 = wmma_bf16(aq[c], load_bfrag_rowk(kr0, lane), s0);
      s1 = wmma_bf16(aq[c], load_bfrag_rowk(kr1, lane), s1);
    }

    // Online softmax (fp32): row r of C-fragment lives in lanes sharing hf.
#pragma unroll
    for (int r = 0; r < 8; ++r) {
      const int qi = qrow + r + 8 * hf;
      float x0 = (kb + ln <= qi)      ? s0[r] * scale : -3.0e38f;
      float x1 = (kb + 16 + ln <= qi) ? s1[r] * scale : -3.0e38f;
      float mx = fmaxf(x0, x1);
      for (int off = 1; off < 16; off <<= 1) mx = fmaxf(mx, __shfl_xor(mx, off, 32));
      const float mnew = fmaxf(mrow[r], mx);
      const float p0 = __expf(x0 - mnew);
      const float p1 = __expf(x1 - mnew);
      float ps = p0 + p1;
      for (int off = 1; off < 16; off <<= 1) ps += __shfl_xor(ps, off, 32);
      const float alpha = __expf(mrow[r] - mnew);
      lrow[r] = lrow[r] * alpha + ps;
      mrow[r] = mnew;
#pragma unroll
      for (int s = 0; s < 8; ++s) oa[s][r] *= alpha;
      // stage P (16x32) in wave-private LDS for A-fragment re-layout
      Pb[wave][(r + 8 * hf) * 32 + ln]      = (bf16)p0;
      Pb[wave][(r + 8 * hf) * 32 + 16 + ln] = (bf16)p1;
    }
    __asm__ volatile("s_wait_dscnt 0x0" ::: "memory");  // wave-local LDS RAW

    // O += P * V : one A-fragment (16x32), 8 DH column slabs of 16
    v16bf ap = load_afrag(&Pb[wave][0], 32, lane);
#pragma unroll
    for (int s = 0; s < 8; ++s)
      oa[s] = wmma_bf16(ap, load_bfrag_rowk(Vtt + (s * 16 + ln) * 32, lane), oa[s]);
  }

  bf16* og = o + ((size_t)(b * SEQ + qrow)) * DMODEL + h * DH;
#pragma unroll
  for (int s = 0; s < 8; ++s)
#pragma unroll
    for (int r = 0; r < 8; ++r) {
      const int row = r + 8 * hf;
      og[(size_t)row * DMODEL + s * 16 + ln] = (bf16)(oa[s][r] / lrow[r]);
    }
}

// ---------------- Elementwise / reduction kernels ---------------------------------

__global__ __launch_bounds__(256) void add_rmsnorm(const float* __restrict__ x,
                                                   const float* __restrict__ t,
                                                   const float* __restrict__ w,
                                                   float* __restrict__ y,
                                                   bf16* __restrict__ yb) {
  const int row = blockIdx.x, tid = threadIdx.x;
  __shared__ float red[256];
  const float* xr = x + (size_t)row * DMODEL;
  const float* tr = t + (size_t)row * DMODEL;
  float ss = 0.f;
  for (int d = tid; d < DMODEL; d += 256) {
    float u = xr[d] + tr[d];
    ss += u * u;
  }
  red[tid] = ss;
  __syncthreads();
  for (int s = 128; s > 0; s >>= 1) {
    if (tid < s) red[tid] += red[tid + s];
    __syncthreads();
  }
  const float inv = rsqrtf(red[0] / (float)DMODEL + 1e-5f);
  for (int d = tid; d < DMODEL; d += 256) {
    float u = (xr[d] + tr[d]) * inv * w[d];
    y[(size_t)row * DMODEL + d] = u;
    if (yb) yb[(size_t)row * DMODEL + d] = (bf16)u;
  }
}

__global__ __launch_bounds__(256) void router_kernel(const float* __restrict__ x1,
                                                     const float* __restrict__ Wr,
                                                     int* __restrict__ counts,
                                                     int* __restrict__ etok,
                                                     float* __restrict__ egate) {
  const int tok = blockIdx.x, tid = threadIdx.x;
  __shared__ float red[NE][256];
  float acc[NE];
#pragma unroll
  for (int e = 0; e < NE; ++e) acc[e] = 0.f;
  const float* xr = x1 + (size_t)tok * DMODEL;
  for (int d = tid; d < DMODEL; d += 256) {
    const float xv = xr[d];
#pragma unroll
    for (int e = 0; e < NE; ++e) acc[e] += xv * Wr[(size_t)d * NE + e];
  }
#pragma unroll
  for (int e = 0; e < NE; ++e) red[e][tid] = acc[e];
  __syncthreads();
  for (int s = 128; s > 0; s >>= 1) {
    if (tid < s) {
#pragma unroll
      for (int e = 0; e < NE; ++e) red[e][tid] += red[e][tid + s];
    }
    __syncthreads();
  }
  if (tid == 0) {
    float m = red[0][0];
    for (int e = 1; e < NE; ++e) m = fmaxf(m, red[e][0]);
    float p[NE];
    for (int e = 0; e < NE; ++e) p[e] = __expf(red[e][0] - m);
    int i1 = 0;
    for (int e = 1; e < NE; ++e)
      if (p[e] > p[i1]) i1 = e;
    int i2 = (i1 == 0) ? 1 : 0;
    for (int e = 0; e < NE; ++e)
      if (e != i1 && p[e] > p[i2]) i2 = e;
    float w1 = p[i1], w2 = p[i2];
    const float sw = w1 + w2;
    w1 /= sw;
    w2 /= sw;
    const int p1 = atomicAdd(&counts[i1], 1);
    etok[i1 * CAP + p1]  = tok;
    egate[i1 * CAP + p1] = w1;
    const int p2 = atomicAdd(&counts[i2], 1);
    etok[i2 * CAP + p2]  = tok;
    egate[i2 * CAP + p2] = w2;
  }
}

__global__ void zero_f32(float* p, int n) {
  const int i = blockIdx.x * 256 + threadIdx.x;
  if (i < n) p[i] = 0.f;
}

__global__ void f32_to_bf16(const float* __restrict__ s, bf16* __restrict__ d, int n) {
  const int i = blockIdx.x * 256 + threadIdx.x;
  if (i < n) d[i] = (bf16)s[i];
}

// ---------------- Launch ----------------------------------------------------------

extern "C" void kernel_launch(void* const* d_in, const int* in_sizes, int n_in,
                              void* d_out, int out_size, void* d_ws, size_t ws_size,
                              hipStream_t stream) {
  (void)in_sizes; (void)n_in; (void)out_size; (void)ws_size;
  const float* x    = (const float*)d_in[0];
  const float* Wdq  = (const float*)d_in[1];
  const float* Wuq  = (const float*)d_in[2];
  const float* Wdkv = (const float*)d_in[3];
  const float* Wuk  = (const float*)d_in[4];
  const float* Wuv  = (const float*)d_in[5];
  const float* Wo   = (const float*)d_in[6];
  const float* n1w  = (const float*)d_in[7];
  const float* Wr   = (const float*)d_in[8];
  const float* We1  = (const float*)d_in[9];
  const float* We2  = (const float*)d_in[10];
  const float* n2w  = (const float*)d_in[11];

  char* wsp  = (char*)d_ws;
  size_t off = 0;
  auto alloc = [&](size_t bytes) -> void* {
    void* p = (void*)(wsp + off);
    off += (bytes + 255) & ~(size_t)255;
    return p;
  };
  bf16*  xb    = (bf16*)alloc((size_t)NT * DMODEL * 2);
  bf16*  ckvb  = (bf16*)alloc((size_t)NT * RLAT * 2);
  bf16*  qlatb = (bf16*)alloc((size_t)NT * RLAT * 2);
  bf16*  qb    = (bf16*)alloc((size_t)NT * DMODEL * 2);
  bf16*  kbuf  = (bf16*)alloc((size_t)NT * DMODEL * 2);
  bf16*  vbuf  = (bf16*)alloc((size_t)NT * DMODEL * 2);
  bf16*  ob    = (bf16*)alloc((size_t)NT * DMODEL * 2);
  float* aproj = (float*)alloc((size_t)NT * DMODEL * 4);
  float* x1    = (float*)alloc((size_t)NT * DMODEL * 4);
  bf16*  x1b   = (bf16*)alloc((size_t)NT * DMODEL * 2);
  float* macc  = (float*)alloc((size_t)NT * DMODEL * 4);
  int*   counts = (int*)alloc(NE * sizeof(int));
  int*   etok   = (int*)alloc((size_t)NE * CAP * sizeof(int));
  float* egate  = (float*)alloc((size_t)NE * CAP * sizeof(float));
  bf16*  hbuf   = (bf16*)alloc((size_t)NE * CAP * HID * 2);

  const int n = NT * DMODEL;
  zero_f32<<<(n + 255) / 256, 256, 0, stream>>>(macc, n);
  zero_f32<<<1, 256, 0, stream>>>((float*)counts, NE);
  f32_to_bf16<<<(n + 255) / 256, 256, 0, stream>>>(x, xb, n);

  // MLA projections (bf16 WMMA, fp32 accum, bf16 intermediates)
  gemm_wmma<true, false, false, false><<<dim3(NT / 128, RLAT / 64, 1), 256, 0, stream>>>(
      xb, Wdkv, ckvb, NT, RLAT, DMODEL, DMODEL, RLAT, nullptr, nullptr, nullptr);
  gemm_wmma<true, false, false, false><<<dim3(NT / 128, RLAT / 64, 1), 256, 0, stream>>>(
      xb, Wdq, qlatb, NT, RLAT, DMODEL, DMODEL, RLAT, nullptr, nullptr, nullptr);
  gemm_wmma<true, false, false, false><<<dim3(NT / 128, DMODEL / 64, 1), 256, 0, stream>>>(
      qlatb, Wuq, qb, NT, DMODEL, RLAT, RLAT, DMODEL, nullptr, nullptr, nullptr);
  gemm_wmma<true, false, false, false><<<dim3(NT / 128, DMODEL / 64, 1), 256, 0, stream>>>(
      ckvb, Wuk, kbuf, NT, DMODEL, RLAT, RLAT, DMODEL, nullptr, nullptr, nullptr);
  gemm_wmma<true, false, false, false><<<dim3(NT / 128, DMODEL / 64, 1), 256, 0, stream>>>(
      ckvb, Wuv, vbuf, NT, DMODEL, RLAT, RLAT, DMODEL, nullptr, nullptr, nullptr);

  attn_wmma<<<dim3(SEQ / 64, BATCH * NH, 1), 128, 0, stream>>>(qb, kbuf, vbuf, ob);

  gemm_wmma<false, false, false, false><<<dim3(NT / 128, DMODEL / 64, 1), 256, 0, stream>>>(
      ob, Wo, aproj, NT, DMODEL, DMODEL, DMODEL, DMODEL, nullptr, nullptr, nullptr);

  add_rmsnorm<<<NT, 256, 0, stream>>>(x, aproj, n1w, x1, x1b);
  router_kernel<<<NT, 256, 0, stream>>>(x1, Wr, counts, etok, egate);

  // MoE: only routed tokens (top-2) hit the expert GEMMs.
  gemm_wmma<true, true, true, false><<<dim3(CAP / 128, HID / 64, NE), 256, 0, stream>>>(
      x1b, We1, hbuf, CAP, HID, DMODEL, DMODEL, HID, etok, nullptr, counts);
  gemm_wmma<false, false, false, true><<<dim3(CAP / 128, DMODEL / 64, NE), 256, 0, stream>>>(
      hbuf, We2, macc, CAP, DMODEL, HID, HID, DMODEL, etok, egate, counts);

  add_rmsnorm<<<NT, 256, 0, stream>>>(x1, macc, n2w, (float*)d_out, nullptr);
}
